// ParityChannel_86260123173234
// MI455X (gfx1250) — compile-verified, matrix operand order checked by reading, not run on hardware
//
#include <hip/hip_runtime.h>
#include <hip/hip_bf16.h>

typedef __attribute__((ext_vector_type(16))) _Float16 v16h;
typedef __attribute__((ext_vector_type(8)))  _Float16 v8h;
typedef __attribute__((ext_vector_type(8)))  float    v8f;

#define BATCH 4
#define N_D   32768
#define N_E   65536
#define EDG   262144
#define H     128
#define ALPHA 0.1f
#define WSTRIDE 136   // padded LDS row stride (halves): 16B-aligned rows, conflict-free

// ---------------- helpers ----------------

// A-fragment (16x32 f16) from a global f32 row. lane: m = lane&15, khalf = lane>>4.
// elements 0..7  -> K = k0..k0+7   (k0 = kc*32 + khalf*8)
// elements 8..15 -> K = k0+16..k0+23
__device__ inline v16h load_afrag_global(const float* rowp, int k0) {
    const float4* p0 = (const float4*)(rowp + k0);
    const float4* p1 = (const float4*)(rowp + k0 + 16);
    float4 f0 = p0[0], f1 = p0[1], f2 = p1[0], f3 = p1[1];
    v16h a;
    a[0]=(_Float16)f0.x; a[1]=(_Float16)f0.y; a[2]=(_Float16)f0.z; a[3]=(_Float16)f0.w;
    a[4]=(_Float16)f1.x; a[5]=(_Float16)f1.y; a[6]=(_Float16)f1.z; a[7]=(_Float16)f1.w;
    a[8]=(_Float16)f2.x; a[9]=(_Float16)f2.y; a[10]=(_Float16)f2.z; a[11]=(_Float16)f2.w;
    a[12]=(_Float16)f3.x; a[13]=(_Float16)f3.y; a[14]=(_Float16)f3.z; a[15]=(_Float16)f3.w;
    return a;
}

// Same fragment pattern from an LDS row (16B-aligned, stride-padded).
__device__ inline v16h load_frag_lds(const _Float16* rowbase, int k0) {
    v8h lo = *(const v8h*)(rowbase + k0);
    v8h hi = *(const v8h*)(rowbase + k0 + 16);
    v16h a;
#pragma unroll
    for (int e = 0; e < 8; ++e) { a[e] = lo[e]; a[e + 8] = hi[e]; }
    return a;
}

// Stage a 128x128 f32 weight (row-major, leading dim ld) into LDS f16 [n][k].
__device__ inline void load_w_lds(_Float16* wT, const float* w, int ld) {
#pragma unroll 4
    for (int i = threadIdx.x; i < 128 * 128; i += 128) {
        int n = i >> 7, k = i & 127;
        wT[n * WSTRIDE + k] = (_Float16)w[n * ld + k];
    }
}

// ---------------- kernels ----------------

__global__ void zero_kernel(float* p, int n) {
    int i = blockIdx.x * blockDim.x + threadIdx.x;
    if (i < n) p[i] = 0.0f;
}

// llr = hE . w_llr ; pt = clamp(-tanh(llr/2), +-0.999). One wave per (b, e-node).
__global__ void pt_kernel(const float* __restrict__ hE, const float* __restrict__ wllr,
                          float* __restrict__ pt) {
    int gid  = blockIdx.x * blockDim.x + threadIdx.x;
    int wid  = gid >> 5, lane = gid & 31;
    if (wid >= BATCH * N_E) return;
    const float4 v = *(const float4*)(hE + (size_t)wid * H + lane * 4);
    const float4 w = *(const float4*)(wllr + lane * 4);
    float s = v.x * w.x + v.y * w.y + v.z * w.z + v.w * w.w;
#pragma unroll
    for (int off = 16; off > 0; off >>= 1) s += __shfl_xor(s, off, 32);
    if (lane == 0) {
        float p = -tanhf(s * 0.5f);
        p = fminf(fmaxf(p, -0.999f), 0.999f);
        pt[wid] = p;
    }
}

// Scatter log|pt| and negative-sign counts into data nodes. One thread per (edge, b).
__global__ void parity_scatter_kernel(const long long* __restrict__ e2d,
                                      const float* __restrict__ pt,
                                      float* __restrict__ logagg,
                                      float* __restrict__ negcnt) {
    int idx = blockIdx.x * blockDim.x + threadIdx.x;
    if (idx >= EDG * BATCH) return;
    int e = idx % EDG, b = idx / EDG;
    long long es = e2d[e];
    long long dd = e2d[EDG + e];
    float p  = pt[(size_t)b * N_E + es];
    float la = logf(fmaxf(fabsf(p), 1e-8f));
    atomicAdd(&logagg[(size_t)b * N_D + dd], la);
    if (p < 0.0f) atomicAdd(&negcnt[(size_t)b * N_D + dd], 1.0f);
}

// Fused 3-stage WMMA MLP over rows of (B*N_D, 128):
//  h1     = relu(hD @ w1[:, :128].T + parity * w1[:,128] + b1)
//  hD_new = hD + ALPHA * (h1 @ w2.T + b2)        -> d_out (hD region)
//  hD_sb  = hD_new @ w_sb.T                      -> ws    (pre-transform for edge scatter)
__global__ __launch_bounds__(128)
void mlp_kernel(const float* __restrict__ hD,
                const float* __restrict__ w1, const float* __restrict__ b1,
                const float* __restrict__ w2, const float* __restrict__ b2,
                const float* __restrict__ wsb,
                const float* __restrict__ logagg, const float* __restrict__ negcnt,
                float* __restrict__ outD, float* __restrict__ hdsb) {
    __shared__ __align__(16) _Float16 wT[128 * WSTRIDE];
    __shared__ __align__(16) _Float16 stage[4][16 * WSTRIDE];
    __shared__ float b1s[128], b2s[128], w1c[128];

    const int wv = threadIdx.x >> 5, lane = threadIdx.x & 31;
    const int mrow = lane & 15, khalf = lane >> 4;
    const int rowBase = blockIdx.x * 64 + wv * 16;

    if (threadIdx.x < 128) {
        b1s[threadIdx.x] = b1[threadIdx.x];
        b2s[threadIdx.x] = b2[threadIdx.x];
        w1c[threadIdx.x] = w1[threadIdx.x * 129 + 128];  // parity column of w1
    }
    load_w_lds(wT, w1, 129);
    __syncthreads();

    // per-lane parity values for C-tile rows m = r + khalf*8
    float par[8];
#pragma unroll
    for (int r = 0; r < 8; ++r) {
        int row = rowBase + r + khalf * 8;
        float la = logagg[row], ng = negcnt[row];
        float sg = 1.0f - 2.0f * fmodf(ng, 2.0f);
        float p  = sg * __expf(fminf(la, 20.0f));
        par[r] = __builtin_isfinite(p) ? p : 0.0f;
    }

    // ---- stage 1: x @ w1.T (K=128 WMMA + rank-1 parity) ----
    const float* rowp = hD + (size_t)(rowBase + mrow) * H;
    __builtin_prefetch(rowp, 0, 0);
    v16h a[4];
#pragma unroll
    for (int kc = 0; kc < 4; ++kc) a[kc] = load_afrag_global(rowp, kc * 32 + khalf * 8);

    _Float16* st = &stage[wv][0];
#pragma unroll
    for (int nt = 0; nt < 8; ++nt) {
        v8f acc = {};
#pragma unroll
        for (int kc = 0; kc < 4; ++kc) {
            v16h bf = load_frag_lds(&wT[(nt * 16 + mrow) * WSTRIDE], kc * 32 + khalf * 8);
            acc = __builtin_amdgcn_wmma_f32_16x16x32_f16(false, a[kc], false, bf,
                                                         (short)0, acc, false, false);
        }
        int n = nt * 16 + mrow;  // C layout: col = lane&15 + nt*16
#pragma unroll
        for (int r = 0; r < 8; ++r) {
            float v = acc[r] + b1s[n] + par[r] * w1c[n];
            st[(r + khalf * 8) * WSTRIDE + n] = (_Float16)fmaxf(v, 0.0f);
        }
    }
    __syncthreads();

    // ---- stage 2: h1 @ w2.T, residual, emit hD_new ----
    load_w_lds(wT, w2, 128);
    __syncthreads();
#pragma unroll
    for (int kc = 0; kc < 4; ++kc) a[kc] = load_frag_lds(st + mrow * WSTRIDE, kc * 32 + khalf * 8);
    __syncthreads();  // all A-frag reads of `stage` complete before it is overwritten
#pragma unroll
    for (int nt = 0; nt < 8; ++nt) {
        v8f acc = {};
#pragma unroll
        for (int kc = 0; kc < 4; ++kc) {
            v16h bf = load_frag_lds(&wT[(nt * 16 + mrow) * WSTRIDE], kc * 32 + khalf * 8);
            acc = __builtin_amdgcn_wmma_f32_16x16x32_f16(false, a[kc], false, bf,
                                                         (short)0, acc, false, false);
        }
        int n = nt * 16 + mrow;
#pragma unroll
        for (int r = 0; r < 8; ++r) {
            int m = r + khalf * 8;
            size_t off = (size_t)(rowBase + m) * H + n;
            float hn = hD[off] + ALPHA * (acc[r] + b2s[n]);
            outD[off] = hn;
            st[m * WSTRIDE + n] = (_Float16)hn;
        }
    }
    __syncthreads();

    // ---- stage 3: hD_new @ w_sb.T -> hdsb ----
    load_w_lds(wT, wsb, 128);
    __syncthreads();
#pragma unroll
    for (int kc = 0; kc < 4; ++kc) a[kc] = load_frag_lds(st + mrow * WSTRIDE, kc * 32 + khalf * 8);
#pragma unroll
    for (int nt = 0; nt < 8; ++nt) {
        v8f acc = {};
#pragma unroll
        for (int kc = 0; kc < 4; ++kc) {
            v16h bf = load_frag_lds(&wT[(nt * 16 + mrow) * WSTRIDE], kc * 32 + khalf * 8);
            acc = __builtin_amdgcn_wmma_f32_16x16x32_f16(false, a[kc], false, bf,
                                                         (short)0, acc, false, false);
        }
        int n = nt * 16 + mrow;
#pragma unroll
        for (int r = 0; r < 8; ++r)
            hdsb[(size_t)(rowBase + r + khalf * 8) * H + n] = acc[r];
    }
}

__global__ void count_kernel(const long long* __restrict__ d2e, float* __restrict__ cnt) {
    int e = blockIdx.x * blockDim.x + threadIdx.x;
    if (e >= EDG) return;
    long long ed = d2e[EDG + e];
    atomicAdd(&cnt[ed], 1.0f);
}

__global__ void copy_kernel(const float* __restrict__ src, float* __restrict__ dst, long long n4) {
    long long i = (long long)blockIdx.x * blockDim.x + threadIdx.x;
    long long stride = (long long)gridDim.x * blockDim.x;
    const float4* s = (const float4*)src;
    float4* d = (float4*)dst;
    for (; i < n4; i += stride) d[i] = s[i];
}

// hE_new += ALPHA * hD_sb[b, d_src] / count[e_dst]. One wave per (edge, b).
__global__ void he_scatter_kernel(const long long* __restrict__ d2e,
                                  const float* __restrict__ cnt,
                                  const float* __restrict__ hdsb,
                                  float* __restrict__ outE) {
    int gid = blockIdx.x * blockDim.x + threadIdx.x;
    int w = gid >> 5, lane = gid & 31;
    if (w >= EDG * BATCH) return;
    int e = w % EDG, b = w / EDG;
    long long ds = d2e[e];
    long long ed = d2e[EDG + e];
    float sc = ALPHA / fmaxf(cnt[ed], 1.0f);
    const float* src = hdsb + ((size_t)b * N_D + ds) * H + lane * 4;
    __builtin_prefetch(src, 0, 0);
    float4 v = *(const float4*)src;
    float* dst = outE + ((size_t)b * N_E + ed) * H + lane * 4;
    atomicAdd(dst + 0, sc * v.x);
    atomicAdd(dst + 1, sc * v.y);
    atomicAdd(dst + 2, sc * v.z);
    atomicAdd(dst + 3, sc * v.w);
}

extern "C" void kernel_launch(void* const* d_in, const int* in_sizes, int n_in,
                              void* d_out, int out_size, void* d_ws, size_t ws_size,
                              hipStream_t stream) {
    const float*      hD   = (const float*)d_in[0];
    const float*      hE   = (const float*)d_in[1];
    const long long*  e2d  = (const long long*)d_in[2];
    const long long*  d2e  = (const long long*)d_in[3];
    const float*      wllr = (const float*)d_in[4];
    const float*      w1   = (const float*)d_in[5];
    const float*      b1   = (const float*)d_in[6];
    const float*      w2   = (const float*)d_in[7];
    const float*      b2   = (const float*)d_in[8];
    const float*      wsb  = (const float*)d_in[9];

    float* outD = (float*)d_out;
    float* outE = outD + (size_t)BATCH * N_D * H;

    float* ws     = (float*)d_ws;
    float* pt     = ws;                                  // B*N_E
    float* logagg = pt + (size_t)BATCH * N_E;            // B*N_D
    float* negcnt = logagg + (size_t)BATCH * N_D;        // B*N_D
    float* cnt    = negcnt + (size_t)BATCH * N_D;        // N_E
    float* hdsb   = cnt + N_E;                           // B*N_D*H

    // zero the accumulator regions (logagg, negcnt, cnt are contiguous)
    int nz = BATCH * N_D * 2 + N_E;
    zero_kernel<<<(nz + 255) / 256, 256, 0, stream>>>(logagg, nz);

    pt_kernel<<<(BATCH * N_E) / 8, 256, 0, stream>>>(hE, wllr, pt);

    parity_scatter_kernel<<<(EDG * BATCH) / 256, 256, 0, stream>>>(e2d, pt, logagg, negcnt);

    mlp_kernel<<<(BATCH * N_D) / 64, 128, 0, stream>>>(hD, w1, b1, w2, b2, wsb,
                                                       logagg, negcnt, outD, hdsb);

    count_kernel<<<EDG / 256, 256, 0, stream>>>(d2e, cnt);

    long long n4 = (long long)BATCH * N_E * H / 4;
    copy_kernel<<<4096, 256, 0, stream>>>(hE, outE, n4);

    he_scatter_kernel<<<(EDG * BATCH * 32) / 256, 256, 0, stream>>>(d2e, cnt, hdsb, outE);
}
